// equivariant_encoder_5463198401071
// MI455X (gfx1250) — compile-verified, matrix operand order checked by reading, not run on hardware
//
#include <hip/hip_runtime.h>
#include <hip/hip_bf16.h>
#include <cstdint>

// ---------------------------------------------------------------------------
// Fused equivariant encoder for MI455X (gfx1250), wave32 + WMMA f16.
//   tile = 16 nodes -> M dimension of v_wmma_f32_16x16x32_f16
//   8 waves / block, wave w owns output columns [16w, 16w+16)
//   All weight B-fragments are loop-invariant per wave -> held in VGPRs
//   (22 fragments x 8 VGPRs = 176 VGPRs), loaded once from global (L2).
//   LDS holds only the 16-node activation ping-pong buffers (~25 KB).
// ---------------------------------------------------------------------------

typedef __attribute__((ext_vector_type(16))) _Float16 v16h;
typedef __attribute__((ext_vector_type(8)))  float    v8f;

#define TPB   256
#define RDIM  128   // RD
#define NRAD  32    // NR
#define ATDIM 32    // ATD
#define EDIM  384   // ED
#define ASTR  168   // activation LDS row stride (halves), conflict-staggered
#define TSTR  132   // tbuf LDS row stride (floats)

// Branch-free tanh-GELU: tanh(u) = 1 - 2/(exp(2u)+1); exact at +/-inf limits.
__device__ __forceinline__ float gelu_f(float x) {
  const float k0 = 0.7978845608028654f;                 // sqrt(2/pi)
  float u = k0 * (x + 0.044715f * x * x * x);
  float e = __expf(2.0f * u);                           // v_exp_f32 path
  float t = 1.0f - 2.0f * __builtin_amdgcn_rcpf(e + 1.0f);
  return 0.5f * x * (1.0f + t);
}

// Build one 16-half WMMA A-fragment from LDS (8 x ds_load_b32).
// dlo = dword offset of this half-wave's K slice (0 for lanes 0-15, 4 for 16-31)
__device__ __forceinline__ v16h ldfrag(const _Float16* base, int dlo) {
  union { v16h h; unsigned u[8]; } f;
  const unsigned* p = (const unsigned*)base;
#pragma unroll
  for (int j = 0; j < 4; ++j) { f.u[j] = p[dlo + j]; f.u[4 + j] = p[dlo + 8 + j]; }
  return f.h;
}

// Load one B-fragment (K=32 block x 16 cols) for this lane straight from a
// row-major global f32 weight block W[32][128], converting to f16.
// Lane layout mirrors the ISA 16-bit 32x16 B matrix striping.
__device__ __forceinline__ v16h ldfrag_g(const float* __restrict__ Wblk, int col,
                                         int lane) {
  const int lo = (lane < 16) ? 0 : 8;
  union { v16h h; _Float16 e[16]; } f;
#pragma unroll
  for (int j = 0; j < 8; ++j) {
    int k = lo + ((j < 4) ? 2 * j : 16 + 2 * (j - 4));
    f.e[2 * j]     = (_Float16)Wblk[(k)     * 128 + col];
    f.e[2 * j + 1] = (_Float16)Wblk[(k + 1) * 128 + col];
  }
  return f.h;
}

// One 16x16 output tile of A[16xK] @ B (B fragments pre-loaded in registers).
template <int K>
__device__ __forceinline__ v8f wgemm(const _Float16* A, const v16h* Bf, int lane) {
  v8f c = {0.f, 0.f, 0.f, 0.f, 0.f, 0.f, 0.f, 0.f};
  const int m   = lane & 15;
  const int dlo = (lane < 16) ? 0 : 4;
#pragma unroll
  for (int kb = 0, i = 0; kb < K; kb += 32, ++i) {
    v16h a = ldfrag(A + m * ASTR + kb, dlo);
    c = __builtin_amdgcn_wmma_f32_16x16x32_f16(false, a, false, Bf[i],
                                               (short)0, c, false, false);
  }
  return c;
}

__global__ void __launch_bounds__(TPB)
enc_zero_ws(float* ws, int n) {
  int i = blockIdx.x * blockDim.x + threadIdx.x;
  if (i < n) ws[i] = 0.0f;
}

__global__ void __launch_bounds__(TPB)
enc_main(const float* __restrict__ pos, const int* __restrict__ zt,
         const int* __restrict__ batch, const float* __restrict__ node_table,
         const float* __restrict__ w_r1, const float* __restrict__ b_r1,
         const float* __restrict__ w_r2, const float* __restrict__ b_r2,
         const float* __restrict__ w_m1, const float* __restrict__ b_m1,
         const float* __restrict__ w_m2, const float* __restrict__ b_m2,
         const float* __restrict__ w_m3, const float* __restrict__ b_m3,
         const float* __restrict__ w_tp,
         float* __restrict__ sums, float* __restrict__ cnt,
         int Nn, int numTiles) {
  __shared__ _Float16 bufA[16 * ASTR];
  __shared__ _Float16 bufB[16 * ASTR];
  __shared__ _Float16 hbuf[16 * ASTR];
  __shared__ float    tbuf[16 * TSTR];
  __shared__ float    shv[16 * 4];
  __shared__ float    distL[16];
  __shared__ int      gidx[16];
  __shared__ int      zL[16];

  const int tid  = threadIdx.x;
  const int lane = tid & 31;
  const int wave = tid >> 5;
  const int colc = (wave << 4) + (lane & 15);
  const int mb   = (lane >> 4) * 8;

  // ---- hoist all loop-invariant B fragments + biases into registers ----
  v16h fr1[1], fr2[4], fm1[5], fm2[4], fm3[4], ftp[4];
  fr1[0] = ldfrag_g(w_r1, colc, lane);
#pragma unroll
  for (int i = 0; i < 4; ++i) fr2[i] = ldfrag_g(w_r2 + 32 * i * 128, colc, lane);
#pragma unroll
  for (int i = 0; i < 5; ++i) fm1[i] = ldfrag_g(w_m1 + 32 * i * 128, colc, lane);
#pragma unroll
  for (int i = 0; i < 4; ++i) fm2[i] = ldfrag_g(w_m2 + 32 * i * 128, colc, lane);
#pragma unroll
  for (int i = 0; i < 4; ++i) fm3[i] = ldfrag_g(w_m3 + 32 * i * 128, colc, lane);
#pragma unroll
  for (int i = 0; i < 4; ++i) ftp[i] = ldfrag_g(w_tp + 32 * i * 128, colc, lane);
  const float bb_r1 = b_r1[colc];
  const float bb_r2 = b_r2[colc];
  const float bb_m1 = b_m1[colc];
  const float bb_m2 = b_m2[colc];
  const float bb_m3 = b_m3[colc];

  for (int tile = blockIdx.x; tile < numTiles; tile += gridDim.x) {
    const int base = tile * 16;

    // ---- per-node scalars: dist, SH(l=1, component norm), graph id, type ----
    if (tid < 16) {
      int node = base + tid;
      bool val = node < Nn;
      int  nc  = val ? node : 0;
      float px = pos[nc * 3 + 0], py = pos[nc * 3 + 1], pz = pos[nc * 3 + 2];
      distL[tid] = sqrtf(px * px + py * py + pz * pz);
      const float s3 = 1.7320508075688772f;
      shv[tid * 4 + 0] = val ? s3 * py : 0.f;   // (y, z, x) order
      shv[tid * 4 + 1] = val ? s3 * pz : 0.f;
      shv[tid * 4 + 2] = val ? s3 * px : 0.f;
      gidx[tid] = val ? batch[nc] : -1;
      zL[tid]   = val ? zt[nc] : 0;
    }
    __syncthreads();

    // ---- gaussian RBF into bufA[16x32]; atom embedding into hbuf cols 128..159
    {
      const float delta = 1.0f / 31.0f;
      const float coeff = -0.5f / (delta * delta);
      for (int v = tid; v < 16 * NRAD; v += TPB) {
        int m = v >> 5, j = v & 31;
        float x = distL[m] - (float)j * delta;
        bufA[m * ASTR + j]        = (_Float16)__expf(coeff * x * x);
        hbuf[m * ASTR + RDIM + j] = (_Float16)node_table[zL[m] * ATDIM + j];
      }
    }
    __syncthreads();

    // ---- GEMM1: gelu(rbf @ w_r1 + b_r1) -> bufB ----
    {
      v8f c = wgemm<32>(bufA, fr1, lane);
#pragma unroll
      for (int r = 0; r < 8; ++r)
        bufB[(mb + r) * ASTR + colc] = (_Float16)gelu_f(c[r] + bb_r1);
    }
    __syncthreads();

    // ---- GEMM2: r_emb = a1 @ w_r2 + b_r2 -> hbuf cols 0..127 ----
    {
      v8f c = wgemm<128>(bufB, fr2, lane);
#pragma unroll
      for (int r = 0; r < 8; ++r)
        hbuf[(mb + r) * ASTR + colc] = (_Float16)(c[r] + bb_r2);
    }
    __syncthreads();

    // ---- GEMM3: gelu([r_emb|nemb] @ w_m1 + b_m1) -> bufA ----
    {
      v8f c = wgemm<160>(hbuf, fm1, lane);
#pragma unroll
      for (int r = 0; r < 8; ++r)
        bufA[(mb + r) * ASTR + colc] = (_Float16)gelu_f(c[r] + bb_m1);
    }
    __syncthreads();

    // ---- GEMM4: gelu(h1 @ w_m2 + b_m2) -> bufB ----
    {
      v8f c = wgemm<128>(bufA, fm2, lane);
#pragma unroll
      for (int r = 0; r < 8; ++r)
        bufB[(mb + r) * ASTR + colc] = (_Float16)gelu_f(c[r] + bb_m2);
    }
    __syncthreads();

    // ---- GEMM5: msg = h2 @ w_m3 + b_m3 -> bufA ----
    {
      v8f c = wgemm<128>(bufB, fm3, lane);
#pragma unroll
      for (int r = 0; r < 8; ++r)
        bufA[(mb + r) * ASTR + colc] = (_Float16)(c[r] + bb_m3);
    }
    __syncthreads();

    // ---- GEMM6: t = (msg @ w_tp) / sqrt(128) -> tbuf (f32) ----
    {
      const float pscale = 0.08838834764831845f;   // 1/sqrt(128)
      v8f c = wgemm<128>(bufA, ftp, lane);
#pragma unroll
      for (int r = 0; r < 8; ++r)
        tbuf[(mb + r) * TSTR + colc] = c[r] * pscale;
    }
    __syncthreads();

    // ---- pooling: c_emb[m, 3u+d] = t[m,u]*sh[m,d]; segment-sum by graph ----
    {
      int g0 = gidx[0], gF = gidx[15];
      if (g0 == gF && g0 >= 0) {                     // common case: one graph
        for (int cc = tid; cc < EDIM; cc += TPB) {
          int u = cc / 3, d = cc - 3 * u;
          float s = 0.f;
#pragma unroll
          for (int m = 0; m < 16; ++m) s += tbuf[m * TSTR + u] * shv[m * 4 + d];
          atomicAdd(&sums[(size_t)g0 * EDIM + cc], s);
        }
        if (tid == 0) atomicAdd(&cnt[g0], 16.0f);
      } else {                                       // tile straddles graphs
        for (int cc = tid; cc < EDIM; cc += TPB) {
          int u = cc / 3, d = cc - 3 * u;
          for (int m = 0; m < 16; ++m) {
            int g = gidx[m];
            if (g >= 0)
              atomicAdd(&sums[(size_t)g * EDIM + cc],
                        tbuf[m * TSTR + u] * shv[m * 4 + d]);
          }
        }
        if (tid < 16 && gidx[tid] >= 0) atomicAdd(&cnt[gidx[tid]], 1.0f);
      }
    }
    __syncthreads();
  }
}

__global__ void __launch_bounds__(TPB)
enc_finalize(float* __restrict__ out, const float* __restrict__ sums,
             const float* __restrict__ cnt, int ng) {
  int i = blockIdx.x * blockDim.x + threadIdx.x;
  if (i < ng * EDIM) {
    int g = i / EDIM;
    float c = fmaxf(cnt[g], 1.0f);
    out[i] = sums[i] * (1.0f + 1.0f / c);   // mean + sum
  }
}

extern "C" void kernel_launch(void* const* d_in, const int* in_sizes, int n_in,
                              void* d_out, int out_size, void* d_ws, size_t ws_size,
                              hipStream_t stream) {
  const float* pos        = (const float*)d_in[0];
  const int*   z          = (const int*)  d_in[1];
  const int*   batch      = (const int*)  d_in[2];
  // d_in[3] = num_graphs scalar (derived from out_size instead)
  const float* node_table = (const float*)d_in[4];
  const float* w_r1 = (const float*)d_in[5];
  const float* b_r1 = (const float*)d_in[6];
  const float* w_r2 = (const float*)d_in[7];
  const float* b_r2 = (const float*)d_in[8];
  const float* w_m1 = (const float*)d_in[9];
  const float* b_m1 = (const float*)d_in[10];
  const float* w_m2 = (const float*)d_in[11];
  const float* b_m2 = (const float*)d_in[12];
  const float* w_m3 = (const float*)d_in[13];
  const float* b_m3 = (const float*)d_in[14];
  const float* w_tp = (const float*)d_in[15];

  const int Nn = in_sizes[0] / 3;
  const int NG = out_size / EDIM;

  float* sums = (float*)d_ws;
  float* cnt  = sums + (size_t)NG * EDIM;

  // zero the accumulators every call (deterministic)
  const int zn = NG * EDIM + NG;
  enc_zero_ws<<<(zn + TPB - 1) / TPB, TPB, 0, stream>>>(sums, zn);

  const int numTiles = (Nn + 15) / 16;
  int blocks = numTiles < 2048 ? numTiles : 2048;

  enc_main<<<blocks, TPB, 0, stream>>>(
      pos, z, batch, node_table,
      w_r1, b_r1, w_r2, b_r2, w_m1, b_m1, w_m2, b_m2, w_m3, b_m3, w_tp,
      sums, cnt, Nn, numTiles);

  enc_finalize<<<(NG * EDIM + TPB - 1) / TPB, TPB, 0, stream>>>(
      (float*)d_out, sums, cnt, NG);
}
